// KANLinear_84619445666117
// MI455X (gfx1250) — matrix operand order repaired and checked by reading, not compiled, not run on hardware
//
#include <hip/hip_runtime.h>

typedef __attribute__((ext_vector_type(16))) __bf16 v16bf;
typedef __attribute__((ext_vector_type(8)))  __bf16 bf16x8;
typedef __attribute__((ext_vector_type(8)))  float  v8f;

#define B_SZ   8192
#define IN_SZ  2048
#define OUT_SZ 2048
#define G_SZ   5
#define SLOT   6
#define EPSF   1e-8f

#define BM 128
#define BN 256
#define CI 16
#define KC (CI*SLOT)      // 96
#define LDK 104           // padded LDS row length (bf16 elems), 208B = 13*16B
#define NCHUNK (IN_SZ/CI) // 128

// coefficient table layout (32 floats per input channel)
#define CF_G  0
#define CF_A1 5
#define CF_B1 9
#define CF_S1 13
#define CF_A2 17
#define CF_B2 20
#define CF_S2 23
#define CF_A3 26
#define CF_B3 28
#define CF_S3 30

union FragA { v16bf v; bf16x8 h[2]; };
union ColU  { __bf16 h[KC]; uint4 q[12]; };

static __device__ __forceinline__ float frcp(float v) {
  return __builtin_amdgcn_rcpf(v);
}
static __device__ __forceinline__ unsigned pk2(float a, float b) {
  union { __bf16 h[2]; unsigned u; } t;
  t.h[0] = (__bf16)a; t.h[1] = (__bf16)b;
  return t.u;
}
static __device__ __forceinline__ int mn4(int v) { return v > 4 ? 4 : v; }

__global__ __launch_bounds__(256) void kan_fused_wmma(
    const float* __restrict__ x,
    const float* __restrict__ base_w,
    const float* __restrict__ base_b,
    const float* __restrict__ spline_w,
    const float* __restrict__ grid,
    float* __restrict__ out)
{
  __shared__ __align__(16) __bf16 As[BM * LDK];     // 26,624 B
  __shared__ __align__(16) __bf16 Bs[BN * LDK];     // 53,248 B
  __shared__ float Coef[CI * 32];                   //  2,048 B

  const int tid  = threadIdx.x;
  const int lane = tid & 31;
  const int w    = tid >> 5;   // wave id 0..7
  const int wRow = w >> 2;     // 0..1  -> 64 rows each
  const int wCol = w & 3;      // 0..3  -> 64 cols each
  const int half = lane >> 4;  // 0/1
  const int l16  = lane & 15;

  const int blockN = blockIdx.x * BN;
  const int blockM = blockIdx.y * BM;

  v8f acc[4][4];
  #pragma unroll
  for (int mt = 0; mt < 4; ++mt)
    #pragma unroll
    for (int nt = 0; nt < 4; ++nt)
      #pragma unroll
      for (int r = 0; r < 8; ++r) acc[mt][nt][r] = 0.0f;

  // per-thread B-fill pointers: one output column per thread
  const int oL = tid;                         // 0..255
  const size_t oG = (size_t)(blockN + oL);
  const float* bw  = base_w  + oG * IN_SZ;
  const float* swp = spline_w + oG * IN_SZ * G_SZ;

  for (int ci = 0; ci < NCHUNK; ++ci) {
    const int iBase = ci * CI;
    __syncthreads();   // previous WMMA phase done reading LDS

    // ---- per-channel spline coefficients (threads 0..15), one i each ----
    if (tid < CI) {
      const float* gp = grid + (size_t)(iBase + tid) * G_SZ;
      float g[5];
      #pragma unroll
      for (int t = 0; t < 5; ++t) g[t] = gp[t];
      float* c = &Coef[tid * 32];
      #pragma unroll
      for (int t = 0; t < 5; ++t) c[CF_G + t] = g[t];
      #pragma unroll
      for (int j = 0; j < 4; ++j) {            // order 1: i2=j+1, i3=min(j+2,4)
        int i3 = mn4(j + 2);
        c[CF_A1 + j] = frcp(g[j + 1] - g[j] + EPSF);
        c[CF_B1 + j] = frcp(g[i3] - g[j + 1] + EPSF);
        c[CF_S1 + j] = g[i3] + g[j];
      }
      #pragma unroll
      for (int j = 0; j < 3; ++j) {            // order 2: i2=min(j+2,4), i3=min(j+3,4)
        int i2 = mn4(j + 2), i3 = mn4(j + 3);
        c[CF_A2 + j] = frcp(g[i2] - g[j] + EPSF);
        c[CF_B2 + j] = frcp(g[i3] - g[j + 1] + EPSF);
        c[CF_S2 + j] = g[i3] + g[j];
      }
      #pragma unroll
      for (int j = 0; j < 2; ++j) {            // order 3: i2=min(j+3,4), i3=4
        int i2 = mn4(j + 3);
        c[CF_A3 + j] = frcp(g[i2] - g[j] + EPSF);
        c[CF_B3 + j] = frcp(g[4] - g[j + 1] + EPSF);
        c[CF_S3 + j] = g[4] + g[j];
      }
    }

    // ---- B tile: one column (96 k-values) per thread, f32 -> bf16 ----
    {
      ColU col;
      const float* bwc = bw  + iBase;
      const float* swc = swp + (size_t)iBase * G_SZ;
      #pragma unroll
      for (int ii = 0; ii < CI; ++ii) {
        col.h[ii * SLOT + 0] = (__bf16)bwc[ii];
        #pragma unroll
        for (int gg = 0; gg < G_SZ; ++gg)
          col.h[ii * SLOT + 1 + gg] = (__bf16)swc[ii * G_SZ + gg];
      }
      uint4* dst = (uint4*)&Bs[oL * LDK];
      #pragma unroll
      for (int q = 0; q < 12; ++q) dst[q] = col.q[q];
      if (ci + 1 < NCHUNK) {                   // gfx1250 global_prefetch_b8
        __builtin_prefetch(bwc + CI, 0, 3);
        __builtin_prefetch(swc + CI * G_SZ, 0, 3);
      }
    }

    __syncthreads();   // coefficients + B tile visible

    // ---- A tile: x and 5 basis values per (b,i), multiplies only ----
    {
      const int iL = tid & 15;
      const int bG = tid >> 4;
      const float* c = &Coef[iL * 32];
      float gA[5], a1[4], b1[4], s1[4], a2[3], b2[3], s2[3], a3[2], b3[2], s3[2];
      #pragma unroll
      for (int t = 0; t < 5; ++t) gA[t] = c[CF_G + t];
      #pragma unroll
      for (int t = 0; t < 4; ++t) { a1[t]=c[CF_A1+t]; b1[t]=c[CF_B1+t]; s1[t]=c[CF_S1+t]; }
      #pragma unroll
      for (int t = 0; t < 3; ++t) { a2[t]=c[CF_A2+t]; b2[t]=c[CF_B2+t]; s2[t]=c[CF_S2+t]; }
      #pragma unroll
      for (int t = 0; t < 2; ++t) { a3[t]=c[CF_A3+t]; b3[t]=c[CF_B3+t]; s3[t]=c[CF_S3+t]; }

      #pragma unroll
      for (int p = 0; p < 8; ++p) {
        const int row = bG + p * 16;
        const size_t xoff = (size_t)(blockM + row) * IN_SZ + iBase + iL;
        const float xv = x[xoff];
        if (ci + 1 < NCHUNK) __builtin_prefetch(&x[xoff + CI], 0, 3);

        float d0[5];
        #pragma unroll
        for (int gg = 0; gg < 5; ++gg) {
          float df = xv - gA[gg];
          d0[gg] = (df >= 0.0f && df < 1.0f) ? 1.0f : 0.0f;
        }
        float r1[5];
        #pragma unroll
        for (int j = 0; j < 4; ++j)
          r1[j] = (xv - 2.0f*gA[j]) * a1[j] * d0[j] + (s1[j] - xv) * b1[j] * d0[j+1];
        r1[4] = d0[4];
        float r2[4];
        #pragma unroll
        for (int j = 0; j < 3; ++j)
          r2[j] = (xv - 2.0f*gA[j]) * a2[j] * r1[j] + (s2[j] - xv) * b2[j] * r1[j+1];
        r2[3] = r1[3];
        float r3[2];
        #pragma unroll
        for (int j = 0; j < 2; ++j)
          r3[j] = (xv - 2.0f*gA[j]) * a3[j] * r2[j] + (s3[j] - xv) * b3[j] * r2[j+1];

        unsigned* dst = (unsigned*)&As[row * LDK];
        dst[iL * 3 + 0] = pk2(xv,    r3[0]);
        dst[iL * 3 + 1] = pk2(r3[1], r2[2]);
        dst[iL * 3 + 2] = pk2(r1[3], d0[4]);
      }
    }

    __syncthreads();   // A tile visible

    // ---- WMMA: 3 k-steps of 32, 16 tiles (64x64) per wave ----
    #pragma unroll
    for (int ks = 0; ks < 3; ++ks) {
      const int kk = ks * 32;
      FragA af[4], bfrag[4];
      #pragma unroll
      for (int mt = 0; mt < 4; ++mt) {
        // A layout: lanes 0-15 hold K {0..7,16..23}, lanes 16-31 hold K {8..15,24..31}
        const __bf16* p = &As[(wRow*64 + mt*16 + l16) * LDK + kk + 8*half];
        af[mt].h[0] = *(const bf16x8*)(p);
        af[mt].h[1] = *(const bf16x8*)(p + 16);
      }
      #pragma unroll
      for (int nt = 0; nt < 4; ++nt) {
        // B layout: lanes 0-15 hold K {0..7,16..23}, lanes 16-31 hold K {8..15,24..31}
        const __bf16* p = &Bs[(wCol*64 + nt*16 + l16) * LDK + kk + 8*half];
        bfrag[nt].h[0] = *(const bf16x8*)(p);
        bfrag[nt].h[1] = *(const bf16x8*)(p + 16);
      }
      #pragma unroll
      for (int mt = 0; mt < 4; ++mt)
        #pragma unroll
        for (int nt = 0; nt < 4; ++nt)
          acc[mt][nt] = __builtin_amdgcn_wmma_f32_16x16x32_bf16(
              false, af[mt].v, false, bfrag[nt].v,
              (short)0, acc[mt][nt], false, false);
    }
  }

  // ---- epilogue: add bias, store f32 ----
  #pragma unroll
  for (int nt = 0; nt < 4; ++nt) {
    const int gCol = blockN + wCol*64 + nt*16 + l16;
    const float bias = base_b[gCol];
    #pragma unroll
    for (int mt = 0; mt < 4; ++mt) {
      const int rBase = blockM + wRow*64 + mt*16 + 8*half;
      #pragma unroll
      for (int r = 0; r < 8; ++r)
        out[(size_t)(rBase + r) * OUT_SZ + gCol] = acc[mt][nt][r] + bias;
    }
  }
}

extern "C" void kernel_launch(void* const* d_in, const int* in_sizes, int n_in,
                              void* d_out, int out_size, void* d_ws, size_t ws_size,
                              hipStream_t stream) {
  (void)in_sizes; (void)n_in; (void)out_size; (void)d_ws; (void)ws_size;
  const float* x        = (const float*)d_in[0];
  const float* base_w   = (const float*)d_in[1];
  const float* base_b   = (const float*)d_in[2];
  const float* spline_w = (const float*)d_in[3];
  const float* grid     = (const float*)d_in[4];
  float* out = (float*)d_out;

  dim3 grd(OUT_SZ / BN, B_SZ / BM);   // (8, 64) = 512 workgroups, 8 waves each
  kan_fused_wmma<<<grd, 256, 0, stream>>>(x, base_w, base_b, spline_w, grid, out);
}
